// EncoderLayer_18915035972254
// MI455X (gfx1250) — matrix operand order repaired and checked by reading, not compiled, not run on hardware
//
#include <hip/hip_runtime.h>

// ---------------------------------------------------------------------------
// Transformer encoder layer for MI455X (gfx1250, wave32, WMMA bf16 + TDM).
// L=2048, N=2, E=1024, H=16, D=64, F=4096. fp32 in/out, bf16 WMMA GEMMs.
// GEMM tiles staged to LDS by the Tensor Data Mover (tensor_load_to_lds),
// double-buffered with s_wait_tensorcnt.
// ---------------------------------------------------------------------------

constexpr int Ls = 2048, Nb = 2, Ed = 1024, Hh = 16, Dd = 64, Ff = 4096;
constexpr int Tt = Ls * Nb;        // 4096 token rows

typedef __attribute__((ext_vector_type(16))) __bf16 v16bf;
typedef __attribute__((ext_vector_type(8)))  float  v8f;
typedef unsigned int u32x4 __attribute__((ext_vector_type(4)));
typedef int          i32x4 __attribute__((ext_vector_type(4)));
typedef int          i32x8 __attribute__((ext_vector_type(8)));

union Frag { v16bf v; uint4 q[2]; };

__device__ __forceinline__ unsigned short f2bf(float f) {
  unsigned u = __builtin_bit_cast(unsigned, f);
  u += 0x7fffu + ((u >> 16) & 1u);          // round-to-nearest-even
  return (unsigned short)(u >> 16);
}

// --------------------- TDM tile load (global -> LDS) -----------------------
// Loads a rows x 64 bf16 tile whose rows are K elements apart in global
// memory into LDS, padding each 128B row with 16B (stride 72 bf16) to keep
// ds_load_b128 fragment reads bank-conflict free. Tracked by TENSORcnt.
__device__ __forceinline__ void tdm_load_tile(const unsigned short* gptr,
                                              unsigned lds_off, int K,
                                              int rows) {
  unsigned long long ga = (unsigned long long)(uintptr_t)gptr;
  u32x4 g0 = {
      1u,                                          // count=1, user descriptor
      lds_off,                                     // lds_addr (bytes)
      (unsigned)ga,                                // global_addr[31:0]
      (unsigned)((ga >> 32) & 0x01FFFFFFu) | 0x80000000u  // addr[56:32]|type=2
  };
  // group1: data_size=2B(1), pad_enable, pad_interval=32 DW(4), pad_amount=4 DW(3)
  i32x8 g1 = {
      (int)0x07110000u,
      (int)(((unsigned)K & 0xFFFFu) << 16),        // tensor_dim0[15:0]
      (int)((((unsigned)K >> 16) & 0xFFFFu) |      // tensor_dim0[31:16]
            (((unsigned)rows & 0xFFFFu) << 16)),   // tensor_dim1[15:0]
      (int)((((unsigned)rows >> 16) & 0xFFFFu) |   // tensor_dim1[31:16]
            (64u << 16)),                          // tile_dim0 = 64
      128,                                         // tile_dim1 = 128 rows
      K,                                           // tensor_dim0_stride
      0, 0
  };
  i32x4 z4 = {0, 0, 0, 0};
#if __clang_major__ >= 23
  i32x8 z8 = {0, 0, 0, 0, 0, 0, 0, 0};
  __builtin_amdgcn_tensor_load_to_lds(g0, g1, z4, z4, z8, 0);
#else
  __builtin_amdgcn_tensor_load_to_lds(g0, g1, z4, z4, 0);
#endif
}

// --------------------------- fp32 -> bf16 conversion -----------------------
__global__ __launch_bounds__(256) void f32_to_bf16_kernel(
    const float* __restrict__ s, unsigned short* __restrict__ d, int n) {
  int i = blockIdx.x * 256 + threadIdx.x;
  if (i < n) d[i] = f2bf(s[i]);
}

// --------------------------------- LayerNorm -------------------------------
__global__ __launch_bounds__(256) void ln_kernel(
    const float* __restrict__ x, const float* __restrict__ w,
    const float* __restrict__ b, unsigned short* __restrict__ out) {
  int row = blockIdx.x;                       // one token row per block
  const float* xr = x + (size_t)row * Ed;
  int tid = threadIdx.x, lane = tid & 31, wave = tid >> 5;
  float s = 0.f, s2 = 0.f;
#pragma unroll
  for (int i = 0; i < Ed / 256; ++i) {
    float v = xr[tid + i * 256];
    s += v; s2 += v * v;
  }
#pragma unroll
  for (int d = 1; d < 32; d <<= 1) {
    s += __shfl_xor(s, d, 32); s2 += __shfl_xor(s2, d, 32);
  }
  __shared__ float red[16];
  if (lane == 0) { red[wave] = s; red[8 + wave] = s2; }
  __syncthreads();
  float ts = 0.f, ts2 = 0.f;
#pragma unroll
  for (int j = 0; j < 8; ++j) { ts += red[j]; ts2 += red[8 + j]; }
  float mean = ts * (1.f / Ed);
  float var  = ts2 * (1.f / Ed) - mean * mean;
  float inv  = rsqrtf(var + 1e-5f);
  unsigned short* orow = out + (size_t)row * Ed;
#pragma unroll
  for (int i = 0; i < Ed / 256; ++i) {
    int c = tid + i * 256;
    orow[c] = f2bf((xr[c] - mean) * inv * w[c] + b[c]);
  }
}

// ------------------------------ WMMA GEMM ----------------------------------
// C(M,N) = A(M,K) * B(N,K)^T   (A,B bf16 row-major; acc fp32)
// Tiles DMAed into LDS by TDM, double-buffered on TENSORcnt.
#define EP_QK   0   // (acc+bias)*scale -> bf16 scatter [n][h][l][d]
#define EP_VT   1   // (acc+bias)       -> bf16 scatter [n][h][d][l]
#define EP_F32R 2   // acc+bias+resid   -> fp32 row-major
#define EP_GELU 3   // gelu(acc+bias)   -> bf16 row-major

template <int MODE>
__global__ __launch_bounds__(256) void gemm_kernel(
    const unsigned short* __restrict__ A, const unsigned short* __restrict__ B,
    const float* __restrict__ bias, const float* __restrict__ resid,
    void* __restrict__ outp, int M, int Nn, int K, float scale) {
  constexpr int BM = 128, BN = 128, BK = 64, LSTR = BK + 8;  // 144B row stride
  __shared__ unsigned short sA[2][BM * LSTR];
  __shared__ unsigned short sB[2][BN * LSTR];
  int tid = threadIdx.x, lane = tid & 31, wave = tid >> 5;
  int hf = lane >> 4, r = lane & 15;
  int wm = wave & 3, wn = wave >> 2;          // 4x2 wave grid over 128x128
  int bm = blockIdx.y * BM, bn = blockIdx.x * BN;

  v8f acc[2][4];
#pragma unroll
  for (int i = 0; i < 2; ++i)
#pragma unroll
    for (int j = 0; j < 4; ++j) acc[i][j] = {};

  const unsigned short* arow = A + (size_t)bm * K;
  const unsigned short* brow = B + (size_t)bn * K;
  int nk = K / BK;
  if (wave == 0) {                            // prime buffer 0
    tdm_load_tile(arow, (unsigned)(uintptr_t)&sA[0][0], K, BM);
    tdm_load_tile(brow, (unsigned)(uintptr_t)&sB[0][0], K, BN);
  }
  for (int i = 0; i < nk; ++i) {
    int cb = i & 1;
    if (wave == 0) {
      if (i + 1 < nk) {                       // prefetch next tile pair
        tdm_load_tile(arow + (i + 1) * BK,
                      (unsigned)(uintptr_t)&sA[cb ^ 1][0], K, BM);
        tdm_load_tile(brow + (i + 1) * BK,
                      (unsigned)(uintptr_t)&sB[cb ^ 1][0], K, BN);
        __builtin_amdgcn_s_wait_tensorcnt(2); // current pair has landed
      } else {
        __builtin_amdgcn_s_wait_tensorcnt(0);
      }
    }
    __syncthreads();                          // publish buffer cb
#pragma unroll
    for (int kc = 0; kc < 2; ++kc) {          // two K=32 WMMA steps
      Frag af[2], bfg[4];
#pragma unroll
      for (int mi = 0; mi < 2; ++mi) {        // A frag: chunks at hf*8, 16+hf*8
        const unsigned short* p =
            &sA[cb][(wm * 32 + mi * 16 + r) * LSTR + kc * 32];
        af[mi].q[0] = *(const uint4*)(p + hf * 8);
        af[mi].q[1] = *(const uint4*)(p + 16 + hf * 8);
      }
#pragma unroll
      for (int nj = 0; nj < 4; ++nj) {        // B frag: chunks at hf*8, 16+hf*8
        const unsigned short* p =
            &sB[cb][(wn * 64 + nj * 16 + r) * LSTR + kc * 32];
        bfg[nj].q[0] = *(const uint4*)(p + hf * 8);
        bfg[nj].q[1] = *(const uint4*)(p + 16 + hf * 8);
      }
#pragma unroll
      for (int mi = 0; mi < 2; ++mi)
#pragma unroll
        for (int nj = 0; nj < 4; ++nj)
          acc[mi][nj] = __builtin_amdgcn_wmma_f32_16x16x32_bf16(
              false, af[mi].v, false, bfg[nj].v, (short)0, acc[mi][nj],
              false, false);
    }
    __syncthreads();                          // done reading buffer cb
  }

#pragma unroll
  for (int mi = 0; mi < 2; ++mi)
#pragma unroll
    for (int nj = 0; nj < 4; ++nj)
#pragma unroll
      for (int e = 0; e < 8; ++e) {
        int rg = bm + wm * 32 + mi * 16 + hf * 8 + e;
        int cg = bn + wn * 64 + nj * 16 + r;
        float v = acc[mi][nj][e] + bias[cg];
        if (MODE == EP_QK) {
          v *= scale;
          int l = rg >> 1, n = rg & 1, h = cg >> 6, d = cg & 63;
          ((unsigned short*)outp)[((size_t)(n * Hh + h) * Ls + l) * Dd + d] =
              f2bf(v);
        } else if (MODE == EP_VT) {
          int l = rg >> 1, n = rg & 1, h = cg >> 6, d = cg & 63;
          ((unsigned short*)outp)[((size_t)(n * Hh + h) * Dd + d) * Ls + l] =
              f2bf(v);
        } else if (MODE == EP_F32R) {
          size_t idx = (size_t)rg * Nn + cg;
          ((float*)outp)[idx] = v + resid[idx];
        } else {  // EP_GELU, exact erf form
          float g = 0.5f * v * (1.f + erff(v * 0.70710678118654752f));
          ((unsigned short*)outp)[(size_t)rg * Nn + cg] = f2bf(g);
        }
      }
}

// ---------------------------- Flash attention ------------------------------
// Q,K: bf16 [n][h][l][64]; Vt: bf16 [n][h][64][l]; O: bf16 (L*N, 1024)
__global__ __launch_bounds__(256) void attn_kernel(
    const unsigned short* __restrict__ Q, const unsigned short* __restrict__ Kq,
    const unsigned short* __restrict__ Vt, unsigned short* __restrict__ O) {
  constexpr int KSTR = 72, VSTR = 136, PSTR = 136;
  __shared__ unsigned short sK[128 * KSTR];       // 18.0 KB
  __shared__ unsigned short sV[64 * VSTR];        // 17.0 KB
  __shared__ unsigned short sP[8 * 16 * PSTR];    // 34.0 KB (per-wave strips)
  int tid = threadIdx.x, lane = tid & 31, wave = tid >> 5;
  int hf = lane >> 4, r = lane & 15;
  int qblk = blockIdx.x;                          // 16 query blocks of 128
  int nh = blockIdx.y;                            // n*16 + h
  const size_t base = (size_t)nh * Ls * Dd;

  Frag qf[2];                                     // Q tile resident in regs
  {
    const unsigned short* qp =
        Q + base + (size_t)(qblk * 128 + wave * 16 + r) * Dd;
#pragma unroll
    for (int c = 0; c < 2; ++c) {
      qf[c].q[0] = *(const uint4*)(qp + c * 32 + hf * 8);
      qf[c].q[1] = *(const uint4*)(qp + c * 32 + 16 + hf * 8);
    }
  }

  v8f o[4];
#pragma unroll
  for (int j = 0; j < 4; ++j) o[j] = {};
  float mrow[8], lrow[8];
#pragma unroll
  for (int e = 0; e < 8; ++e) { mrow[e] = -3.0e38f; lrow[e] = 0.f; }

  for (int kt = 0; kt <= qblk; ++kt) {            // causal: only lower tiles
#pragma unroll
    for (int i = 0; i < 4; ++i) {                 // K tile 128x64
      int c = tid + i * 256;
      int row = c >> 3, kc = (c & 7) * 8;
      *(uint4*)(&sK[row * KSTR + kc]) =
          *(const uint4*)(Kq + base + (size_t)(kt * 128 + row) * Dd + kc);
    }
#pragma unroll
    for (int i = 0; i < 4; ++i) {                 // Vt tile 64x128
      int c = tid + i * 256;
      int row = c >> 4, kc = (c & 15) * 8;
      *(uint4*)(&sV[row * VSTR + kc]) =
          *(const uint4*)(Vt + base + (size_t)row * Ls + kt * 128 + kc);
    }
    __syncthreads();

    v8f s[8];                                     // 16q x 128k scores
#pragma unroll
    for (int ks = 0; ks < 8; ++ks) {
      const unsigned short* kp = &sK[(ks * 16 + r) * KSTR];
      Frag k0, k1;
      k0.q[0] = *(const uint4*)(kp + hf * 8);
      k0.q[1] = *(const uint4*)(kp + 16 + hf * 8);
      k1.q[0] = *(const uint4*)(kp + 32 + hf * 8);
      k1.q[1] = *(const uint4*)(kp + 48 + hf * 8);
      v8f z = {};
      z = __builtin_amdgcn_wmma_f32_16x16x32_bf16(false, qf[0].v, false, k0.v,
                                                  (short)0, z, false, false);
      s[ks] = __builtin_amdgcn_wmma_f32_16x16x32_bf16(
          false, qf[1].v, false, k1.v, (short)0, z, false, false);
    }
    if (kt == qblk) {                             // diagonal block mask
#pragma unroll
      for (int ks = 0; ks < 8; ++ks)
#pragma unroll
        for (int e = 0; e < 8; ++e) {
          int qr = wave * 16 + hf * 8 + e;
          int kr = ks * 16 + r;
          if (kr > qr) s[ks][e] = -10000.0f;
        }
    }
    // online softmax (per query row; cols live across a 16-lane half)
    float alpha[8];
#pragma unroll
    for (int e = 0; e < 8; ++e) {
      float mx = s[0][e];
#pragma unroll
      for (int ks = 1; ks < 8; ++ks) mx = fmaxf(mx, s[ks][e]);
#pragma unroll
      for (int d = 1; d < 16; d <<= 1) mx = fmaxf(mx, __shfl_xor(mx, d, 32));
      float mn = fmaxf(mrow[e], mx);
      alpha[e] = __expf(mrow[e] - mn);
      mrow[e] = mn;
      float sum = 0.f;
#pragma unroll
      for (int ks = 0; ks < 8; ++ks) {
        float p = __expf(s[ks][e] - mn);
        s[ks][e] = p;
        sum += p;
      }
#pragma unroll
      for (int d = 1; d < 16; d <<= 1) sum += __shfl_xor(sum, d, 32);
      lrow[e] = lrow[e] * alpha[e] + sum;
#pragma unroll
      for (int j = 0; j < 4; ++j) o[j][e] *= alpha[e];
    }
    // C-layout -> A-fragment layout via per-wave LDS strip (DS in-order/wave)
    unsigned short* pw = &sP[wave * 16 * PSTR];
#pragma unroll
    for (int ks = 0; ks < 8; ++ks)
#pragma unroll
      for (int e = 0; e < 8; ++e)
        pw[(hf * 8 + e) * PSTR + ks * 16 + r] = f2bf(s[ks][e]);
    // O += P * V
#pragma unroll
    for (int c = 0; c < 4; ++c) {                 // 32-key chunks
      Frag pf;
      const unsigned short* pr = &sP[(wave * 16 + r) * PSTR + c * 32];
      pf.q[0] = *(const uint4*)(pr + hf * 8);
      pf.q[1] = *(const uint4*)(pr + 16 + hf * 8);
#pragma unroll
      for (int j = 0; j < 4; ++j) {               // 4 d-tiles
        Frag vf;
        const unsigned short* vp = &sV[(j * 16 + r) * VSTR + c * 32];
        vf.q[0] = *(const uint4*)(vp + hf * 8);
        vf.q[1] = *(const uint4*)(vp + 16 + hf * 8);
        o[j] = __builtin_amdgcn_wmma_f32_16x16x32_bf16(
            false, pf.v, false, vf.v, (short)0, o[j], false, false);
      }
    }
    __syncthreads();
  }

  int n = nh >> 4, h = nh & 15;
#pragma unroll
  for (int j = 0; j < 4; ++j)
#pragma unroll
    for (int e = 0; e < 8; ++e) {
      int lq = qblk * 128 + wave * 16 + hf * 8 + e;
      int col = h * 64 + j * 16 + r;
      O[((size_t)lq * Nb + n) * (Hh * Dd) + col] = f2bf(o[j][e] / lrow[e]);
    }
}

// ------------------------------- launcher ----------------------------------
extern "C" void kernel_launch(void* const* d_in, const int* in_sizes, int n_in,
                              void* d_out, int out_size, void* d_ws,
                              size_t ws_size, hipStream_t stream) {
  (void)in_sizes; (void)n_in; (void)out_size; (void)ws_size;
  const float* x    = (const float*)d_in[0];
  const float* ln1w = (const float*)d_in[1];
  const float* ln1b = (const float*)d_in[2];
  const float* wq   = (const float*)d_in[3];
  const float* bq   = (const float*)d_in[4];
  const float* wk   = (const float*)d_in[5];
  const float* bk   = (const float*)d_in[6];
  const float* wv   = (const float*)d_in[7];
  const float* bv   = (const float*)d_in[8];
  const float* wo   = (const float*)d_in[9];
  const float* bo   = (const float*)d_in[10];
  const float* ln2w = (const float*)d_in[11];
  const float* ln2b = (const float*)d_in[12];
  const float* f1w  = (const float*)d_in[13];
  const float* f1b  = (const float*)d_in[14];
  const float* f2w  = (const float*)d_in[15];
  const float* f2b  = (const float*)d_in[16];

  char* cur = (char*)d_ws;
  auto alloc = [&](size_t bytes) {
    char* p = cur;
    cur += (bytes + 255) & ~(size_t)255;
    return p;
  };
  unsigned short* wq_h  = (unsigned short*)alloc((size_t)Ed * Ed * 2);
  unsigned short* wk_h  = (unsigned short*)alloc((size_t)Ed * Ed * 2);
  unsigned short* wv_h  = (unsigned short*)alloc((size_t)Ed * Ed * 2);
  unsigned short* wo_h  = (unsigned short*)alloc((size_t)Ed * Ed * 2);
  unsigned short* f1w_h = (unsigned short*)alloc((size_t)Ff * Ed * 2);
  unsigned short* f2w_h = (unsigned short*)alloc((size_t)Ed * Ff * 2);
  unsigned short* h1    = (unsigned short*)alloc((size_t)Tt * Ed * 2);
  unsigned short* qb_   = (unsigned short*)alloc((size_t)Tt * Ed * 2);
  unsigned short* kb_   = (unsigned short*)alloc((size_t)Tt * Ed * 2);
  unsigned short* vb_   = (unsigned short*)alloc((size_t)Tt * Ed * 2);
  unsigned short* ob_   = (unsigned short*)alloc((size_t)Tt * Ed * 2);
  float*          x1    = (float*)alloc((size_t)Tt * Ed * 4);
  unsigned short* h2    = (unsigned short*)alloc((size_t)Tt * Ed * 2);
  unsigned short* mid   = (unsigned short*)alloc((size_t)Tt * Ff * 2);

  auto cvt = [&](const float* s, unsigned short* d, int n) {
    f32_to_bf16_kernel<<<(n + 255) / 256, 256, 0, stream>>>(s, d, n);
  };
  cvt(wq, wq_h, Ed * Ed);
  cvt(wk, wk_h, Ed * Ed);
  cvt(wv, wv_h, Ed * Ed);
  cvt(wo, wo_h, Ed * Ed);
  cvt(f1w, f1w_h, Ff * Ed);
  cvt(f2w, f2w_h, Ed * Ff);

  ln_kernel<<<Tt, 256, 0, stream>>>(x, ln1w, ln1b, h1);

  dim3 gEE(Ed / 128, Tt / 128);
  gemm_kernel<EP_QK><<<gEE, 256, 0, stream>>>(h1, wq_h, bq, nullptr, qb_,
                                              Tt, Ed, Ed, 0.125f);
  gemm_kernel<EP_QK><<<gEE, 256, 0, stream>>>(h1, wk_h, bk, nullptr, kb_,
                                              Tt, Ed, Ed, 1.0f);
  gemm_kernel<EP_VT><<<gEE, 256, 0, stream>>>(h1, wv_h, bv, nullptr, vb_,
                                              Tt, Ed, Ed, 1.0f);

  attn_kernel<<<dim3(Ls / 128, Nb * Hh), 256, 0, stream>>>(qb_, kb_, vb_, ob_);

  gemm_kernel<EP_F32R><<<gEE, 256, 0, stream>>>(ob_, wo_h, bo, x, x1,
                                                Tt, Ed, Ed, 1.0f);
  ln_kernel<<<Tt, 256, 0, stream>>>(x1, ln2w, ln2b, h2);
  gemm_kernel<EP_GELU><<<dim3(Ff / 128, Tt / 128), 256, 0, stream>>>(
      h2, f1w_h, f1b, nullptr, mid, Tt, Ff, Ed, 1.0f);
  gemm_kernel<EP_F32R><<<gEE, 256, 0, stream>>>(mid, f2w_h, f2b, x1,
                                                (float*)d_out, Tt, Ed, Ff, 1.0f);
}
